// transformer_85426899517955
// MI455X (gfx1250) — matrix operand order repaired and checked by reading, not compiled, not run on hardware
//
#include <hip/hip_runtime.h>
#include <hip/hip_bf16.h>

// ---------------------------------------------------------------------------
// Problem constants (match reference)
// ---------------------------------------------------------------------------
#define BB    2
#define NN    1024
#define DIMC  1024
#define HH    16
#define DD    64
#define DEPTH 4
#define MULT  4
#define PBN   256
#define TOK   (BB * NN)          // 2048 tokens
#define GLUN  (2 * DIMC * MULT)  // 8192
#define FFK   (DIMC * MULT)      // 4096

typedef __attribute__((ext_vector_type(16))) _Float16 v16h;
typedef __attribute__((ext_vector_type(8)))  _Float16 v8h;
typedef __attribute__((ext_vector_type(8)))  float    v8f;
typedef int v4i __attribute__((vector_size(16)));

#if __has_builtin(__builtin_amdgcn_global_load_async_to_lds_b128) && \
    __has_builtin(__builtin_amdgcn_s_wait_asynccnt)
#define HAVE_ASYNC 1
#else
#define HAVE_ASYNC 0
#endif

// ---------------------------------------------------------------------------
// WMMA helpers
// ---------------------------------------------------------------------------
static __device__ inline v8f wmma_f16(v16h a, v16h b, v8f c) {
    // D = A(16x32 f16) * B(32x16 f16) + C(16x16 f32)
    return __builtin_amdgcn_wmma_f32_16x16x32_f16(
        /*neg_a=*/false, a, /*neg_b=*/false, b,
        /*c_mod=*/(short)0, c, /*reuse_a=*/false, /*reuse_b=*/false);
}

static __device__ inline v16h combine16(const _Float16* p0, const _Float16* p1) {
    v8h a = *(const v8h*)p0;
    v8h b = *(const v8h*)p1;
    v16h r;
#pragma unroll
    for (int i = 0; i < 8; ++i) { r[i] = a[i]; r[8 + i] = b[i]; }
    return r;
}

// A fragment: row-major A (MxK, ld=lda), tile at (m0,k0), K-chunk of 32.
static __device__ inline v16h load_a(const _Float16* A, int lda, int m0, int k0, int lane) {
    int r  = m0 + (lane & 15);
    int kk = k0 + ((lane >> 4) << 3);
    const _Float16* base = A + (size_t)r * lda + kk;
    return combine16(base, base + 16);
}

// B fragment from BT (N x K row-major, ld=ldb): cols n0..n0+15, K-chunk 32.
static __device__ inline v16h load_b(const _Float16* BT, int ldb, int n0, int k0, int lane) {
    int n  = n0 + (lane & 15);
    int kk = k0 + ((lane >> 4) << 3);
    const _Float16* base = BT + (size_t)n * ldb + kk;
    return combine16(base, base + 16);
}

static __device__ inline float red_max16(float v) {
#pragma unroll
    for (int m = 1; m <= 8; m <<= 1) v = fmaxf(v, __shfl_xor(v, m, 32));
    return v;
}
static __device__ inline float red_sum16(float v) {
#pragma unroll
    for (int m = 1; m <= 8; m <<= 1) v += __shfl_xor(v, m, 32);
    return v;
}

// 16-byte chunk copy global -> LDS (async on gfx1250, sync fallback)
static __device__ inline void copy16_g2l(const _Float16* g, _Float16* l) {
#if HAVE_ASYNC
    __builtin_amdgcn_global_load_async_to_lds_b128(
        (__attribute__((address_space(1))) v4i*)g,
        (__attribute__((address_space(3))) v4i*)l,
        0, 0);
#else
    *(v8h*)l = *(const v8h*)g;
#endif
}

// ---------------------------------------------------------------------------
// WMMA GEMM with double-buffered async LDS staging.
// C[M x Ncol] = A[M x K](f16) * BT[Ncol x K](f16)^T (+bias[col]) (+res)
// Block: 256 thr = 8 waves, tile 128(M) x 128(N); wave tile 32x64.
// LDS tiles padded to 40 halves/row -> conflict-free ds_load_b128 fragments.
// ---------------------------------------------------------------------------
#define LDP 40   // padded row pitch (halves) for 32-half tile rows

__global__ __launch_bounds__(256) void gemm_f16_f32(
    const _Float16* __restrict__ A, const _Float16* __restrict__ BT,
    float* Cout, const float* __restrict__ bias, const float* res,
    int Ncol, int K)
{
    __shared__ __align__(16) _Float16 Abuf[2][128 * LDP];
    __shared__ __align__(16) _Float16 Bbuf[2][128 * LDP];

    const int tid  = threadIdx.x;
    const int lane = tid & 31;
    const int wave = tid >> 5;
    const int wm = wave >> 1;           // 0..3 -> 32-row strip
    const int wn = wave & 1;            // 0..1 -> 64-col strip
    const int m0 = blockIdx.y * 128;
    const int n0 = blockIdx.x * 128;

    // Stage one 128x32 tile of A and B into LDS buffer `buf`.
    // 512 chunks of 16B per matrix; 2 chunks per thread per matrix.
    auto stage = [&](int buf, int k0) {
#pragma unroll
        for (int j = 0; j < 2; ++j) {
            int c   = tid + j * 256;        // 0..511
            int row = c >> 2;               // 0..127
            int off = (c & 3) * 8;          // halves within row
            copy16_g2l(A  + (size_t)(m0 + row) * K + k0 + off,
                       &Abuf[buf][row * LDP + off]);
            copy16_g2l(BT + (size_t)(n0 + row) * K + k0 + off,
                       &Bbuf[buf][row * LDP + off]);
        }
    };

    v8f acc[2][4];
#pragma unroll
    for (int fm = 0; fm < 2; ++fm)
#pragma unroll
        for (int fn = 0; fn < 4; ++fn) acc[fm][fn] = (v8f){};

    const int ksteps = K >> 5;
    stage(0, 0);
    for (int kt = 0; kt < ksteps; ++kt) {
        int cur = kt & 1;
        bool has_next = (kt + 1) < ksteps;
        if (has_next) stage(cur ^ 1, (kt + 1) << 5);
#if HAVE_ASYNC
        if (has_next) __builtin_amdgcn_s_wait_asynccnt(4);
        else          __builtin_amdgcn_s_wait_asynccnt(0);
#endif
        __syncthreads();

        const _Float16* At = &Abuf[cur][0];
        const _Float16* Bt = &Bbuf[cur][0];
        v16h af[2], bf[4];
#pragma unroll
        for (int fm = 0; fm < 2; ++fm)
            af[fm] = load_a(At, LDP, wm * 32 + fm * 16, 0, lane);
#pragma unroll
        for (int fn = 0; fn < 4; ++fn)
            bf[fn] = load_b(Bt, LDP, wn * 64 + fn * 16, 0, lane);
#pragma unroll
        for (int fm = 0; fm < 2; ++fm)
#pragma unroll
            for (int fn = 0; fn < 4; ++fn)
                acc[fm][fn] = wmma_f16(af[fm], bf[fn], acc[fm][fn]);

        __syncthreads();
    }

    const int hi = (lane >> 4) << 3;
    const int cl = lane & 15;
#pragma unroll
    for (int fm = 0; fm < 2; ++fm) {
#pragma unroll
        for (int fn = 0; fn < 4; ++fn) {
            int c = n0 + wn * 64 + fn * 16 + cl;
            float badd = bias ? bias[c] : 0.0f;
#pragma unroll
            for (int i = 0; i < 8; ++i) {
                int r = m0 + wm * 32 + fm * 16 + hi + i;
                size_t off = (size_t)r * Ncol + c;
                float v = acc[fm][fn][i] + badd;
                if (res) v += res[off];
                Cout[off] = v;
            }
        }
    }
}

// ---------------------------------------------------------------------------
// LayerNorm -> f16 activation
// ---------------------------------------------------------------------------
__global__ __launch_bounds__(256) void ln_kernel(
    const float* __restrict__ x, const float* __restrict__ g,
    const float* __restrict__ bta, _Float16* __restrict__ out)
{
    __shared__ float red[2][8];
    int t = blockIdx.x;
    const float* xp = x + (size_t)t * DIMC;
    float s = 0.f, s2 = 0.f;
    for (int j = threadIdx.x; j < DIMC; j += 256) {
        float v = xp[j]; s += v; s2 += v * v;
    }
#pragma unroll
    for (int m = 1; m <= 16; m <<= 1) { s += __shfl_xor(s, m, 32); s2 += __shfl_xor(s2, m, 32); }
    int wave = threadIdx.x >> 5, lane = threadIdx.x & 31;
    if (lane == 0) { red[0][wave] = s; red[1][wave] = s2; }
    __syncthreads();
    float ts = 0.f, ts2 = 0.f;
#pragma unroll
    for (int w = 0; w < 8; ++w) { ts += red[0][w]; ts2 += red[1][w]; }
    float mu = ts / DIMC;
    float var = ts2 / DIMC - mu * mu;
    float rs = rsqrtf(var + 1e-5f);
    for (int j = threadIdx.x; j < DIMC; j += 256)
        out[(size_t)t * DIMC + j] = (_Float16)((xp[j] - mu) * rs * g[j] + bta[j]);
}

// ---------------------------------------------------------------------------
// QKV pack: l2norm(q)*temp, l2norm(k), V transposed. One wave per (b,h,n).
// ---------------------------------------------------------------------------
__global__ __launch_bounds__(32) void pack_qkv(
    const float* __restrict__ qkv, const float* __restrict__ temp, int layer,
    _Float16* __restrict__ q16, _Float16* __restrict__ k16, _Float16* __restrict__ v16t)
{
    int n = blockIdx.x;
    int bh = blockIdx.y;
    int lane = threadIdx.x;
    const float* base = qkv + ((size_t)((bh >> 4) * NN + n)) * (3 * HH * DD) + (size_t)(bh & 15) * DD * 3;
    float q0 = base[lane * 3 + 0],        k0 = base[lane * 3 + 1],        v0 = base[lane * 3 + 2];
    float q1 = base[(lane + 32) * 3 + 0], k1 = base[(lane + 32) * 3 + 1], v1 = base[(lane + 32) * 3 + 2];
    float nq = q0 * q0 + q1 * q1;
    float nk = k0 * k0 + k1 * k1;
#pragma unroll
    for (int m = 1; m <= 16; m <<= 1) { nq += __shfl_xor(nq, m, 32); nk += __shfl_xor(nk, m, 32); }
    float sq = temp[layer] / fmaxf(sqrtf(nq), 1e-12f);
    float sk = 1.0f / fmaxf(sqrtf(nk), 1e-12f);
    size_t ob = ((size_t)bh * NN + n) * DD;
    q16[ob + lane]      = (_Float16)(q0 * sq);
    q16[ob + lane + 32] = (_Float16)(q1 * sq);
    k16[ob + lane]      = (_Float16)(k0 * sk);
    k16[ob + lane + 32] = (_Float16)(k1 * sk);
    v16t[((size_t)bh * DD + lane) * NN + n]      = (_Float16)v0;
    v16t[((size_t)bh * DD + lane + 32) * NN + n] = (_Float16)v1;
}

// ---------------------------------------------------------------------------
// Flash attention: 1 wave per 16 query rows, causal, +rel-pos bias table.
// ---------------------------------------------------------------------------
__global__ __launch_bounds__(128) void attn_kernel(
    const _Float16* __restrict__ Q, const _Float16* __restrict__ Kd,
    const _Float16* __restrict__ Vt, const float* __restrict__ btab,
    _Float16* __restrict__ O16)
{
    __shared__ __align__(16) _Float16 Plds[4][16 * 32];
    int lane = threadIdx.x & 31;
    int wave = threadIdx.x >> 5;
    int bh = blockIdx.x;
    int b = bh >> 4, h = bh & 15;
    int n0 = blockIdx.y * 64 + wave * 16;

    const _Float16* Qh = Q  + (size_t)bh * NN * DD;
    const _Float16* Kh = Kd + (size_t)bh * NN * DD;
    const _Float16* Vh = Vt + (size_t)bh * DD * NN;

    v16h qf0 = load_a(Qh, DD, n0, 0,  lane);
    v16h qf1 = load_a(Qh, DD, n0, 32, lane);

    v8f o0 = {}, o1 = {}, o2 = {}, o3 = {};
    float mrow[8], lrow[8];
#pragma unroll
    for (int i = 0; i < 8; ++i) { mrow[i] = -1e30f; lrow[i] = 0.f; }

    int hi = (lane >> 4) << 3;
    int cl = lane & 15;
    int mmax = blockIdx.y * 64 + 63;
    _Float16* pl = &Plds[wave][0];

    for (int m0 = 0; m0 <= mmax; m0 += 32) {
        v8f s0 = {}, s1 = {};
        {
            v16h kf0 = load_b(Kh, DD, m0,      0,  lane);
            v16h kf1 = load_b(Kh, DD, m0,      32, lane);
            s0 = wmma_f16(qf0, kf0, s0);
            s0 = wmma_f16(qf1, kf1, s0);
            v16h kf2 = load_b(Kh, DD, m0 + 16, 0,  lane);
            v16h kf3 = load_b(Kh, DD, m0 + 16, 32, lane);
            s1 = wmma_f16(qf0, kf2, s1);
            s1 = wmma_f16(qf1, kf3, s1);
        }
#pragma unroll
        for (int i = 0; i < 8; ++i) {
            int r  = n0 + hi + i;
            int c0 = m0 + cl;
            int c1 = m0 + 16 + cl;
            float v0 = (c0 > r) ? -1e30f : (s0[i] + btab[(size_t)(r - c0 + NN - 1) * HH + h]);
            float v1 = (c1 > r) ? -1e30f : (s1[i] + btab[(size_t)(r - c1 + NN - 1) * HH + h]);
            float nm = fmaxf(mrow[i], red_max16(fmaxf(v0, v1)));
            float al = __expf(mrow[i] - nm);
            float p0 = __expf(v0 - nm);
            float p1 = __expf(v1 - nm);
            float ps = red_sum16(p0 + p1);
            lrow[i] = lrow[i] * al + ps;
            mrow[i] = nm;
            o0[i] *= al; o1[i] *= al; o2[i] *= al; o3[i] *= al;
            pl[(hi + i) * 32 + cl]      = (_Float16)p0;
            pl[(hi + i) * 32 + 16 + cl] = (_Float16)p1;
        }
        // Same-wave LDS ops are in-order; re-fragment P (16x32) as WMMA A.
        v16h pf = load_a(pl, 32, 0, 0, lane);
        v16h vf;
        vf = load_b(Vh, NN, 0,  m0, lane); o0 = wmma_f16(pf, vf, o0);
        vf = load_b(Vh, NN, 16, m0, lane); o1 = wmma_f16(pf, vf, o1);
        vf = load_b(Vh, NN, 32, m0, lane); o2 = wmma_f16(pf, vf, o2);
        vf = load_b(Vh, NN, 48, m0, lane); o3 = wmma_f16(pf, vf, o3);
    }

#pragma unroll
    for (int i = 0; i < 8; ++i) {
        float inv = 1.0f / lrow[i];
        int r = n0 + hi + i;
        size_t ob = ((size_t)(b * NN + r)) * (HH * DD) + (size_t)h * DD;
        O16[ob + cl]      = (_Float16)(o0[i] * inv);
        O16[ob + 16 + cl] = (_Float16)(o1[i] * inv);
        O16[ob + 32 + cl] = (_Float16)(o2[i] * inv);
        O16[ob + 48 + cl] = (_Float16)(o3[i] * inv);
    }
}

// ---------------------------------------------------------------------------
// GLU: g16[t][j] = a * silu(gate)
// ---------------------------------------------------------------------------
__global__ __launch_bounds__(256) void glu_kernel(
    const float* __restrict__ hbuf, _Float16* __restrict__ g16)
{
    size_t idx = (size_t)blockIdx.x * 256 + threadIdx.x;
    if (idx >= (size_t)TOK * FFK) return;
    size_t t = idx >> 12;
    size_t j = idx & 4095;
    const float* hp = hbuf + t * GLUN;
    float a = hp[j];
    float g = hp[FFK + j];
    g16[idx] = (_Float16)(a * g / (1.0f + __expf(-g)));
}

// ---------------------------------------------------------------------------
// Weight convert + transpose: WT(f16, CxK) = W(f32, KxC)^T
// ---------------------------------------------------------------------------
__global__ __launch_bounds__(256) void conv_t_kernel(
    const float* __restrict__ W, _Float16* __restrict__ WT, int K, int C)
{
    size_t idx = (size_t)blockIdx.x * 256 + threadIdx.x;
    if (idx >= (size_t)K * C) return;
    size_t c = idx / (size_t)K;
    size_t k = idx % (size_t)K;
    WT[idx] = (_Float16)W[k * (size_t)C + c];
}

// ---------------------------------------------------------------------------
// Positional-bias MLP: table[r][h], r in [0, 2N-2]
// ---------------------------------------------------------------------------
__global__ __launch_bounds__(PBN) void posbias_kernel(
    const float* __restrict__ w1, const float* __restrict__ b1,
    const float* __restrict__ w2, const float* __restrict__ b2,
    const float* __restrict__ w3, const float* __restrict__ b3,
    float* __restrict__ table)
{
    __shared__ float h1[PBN], h2[PBN];
    int r = blockIdx.x;
    float pos = (float)(r - (NN - 1));
    int j = threadIdx.x;
    h1[j] = fmaxf(pos * w1[j] + b1[j], 0.0f);
    __syncthreads();
    float acc = b2[j];
    for (int i = 0; i < PBN; ++i) acc += h1[i] * w2[i * PBN + j];
    h2[j] = fmaxf(acc, 0.0f);
    __syncthreads();
    if (j < HH) {
        float a = b3[j];
        for (int i = 0; i < PBN; ++i) a += h2[i] * w3[i * HH + j];
        table[(size_t)r * HH + j] = a;
    }
}

// ---------------------------------------------------------------------------
// Host orchestration
// ---------------------------------------------------------------------------
extern "C" void kernel_launch(void* const* d_in, const int* in_sizes, int n_in,
                              void* d_out, int out_size, void* d_ws, size_t ws_size,
                              hipStream_t stream)
{
    const float* x_in  = (const float*)d_in[0];
    const float* ln1_g = (const float*)d_in[2];
    const float* ln1_b = (const float*)d_in[3];
    const float* w_qkv = (const float*)d_in[4];
    const float* w_out = (const float*)d_in[5];
    const float* temp  = (const float*)d_in[6];
    const float* ln2_g = (const float*)d_in[7];
    const float* ln2_b = (const float*)d_in[8];
    const float* w_glu = (const float*)d_in[9];
    const float* b_glu = (const float*)d_in[10];
    const float* w_ff2 = (const float*)d_in[11];
    const float* b_ff2 = (const float*)d_in[12];
    const float* pb_w1 = (const float*)d_in[13];
    const float* pb_b1 = (const float*)d_in[14];
    const float* pb_w2 = (const float*)d_in[15];
    const float* pb_b2 = (const float*)d_in[16];
    const float* pb_w3 = (const float*)d_in[17];
    const float* pb_b3 = (const float*)d_in[18];
    float* x = (float*)d_out;

    char* p = (char*)d_ws;
    auto alloc = [&](size_t bytes) -> void* {
        void* r = (void*)p;
        p += (bytes + 255) & ~(size_t)255;
        return r;
    };
    const size_t QKV_W = (size_t)3 * HH * DD * DIMC;
    const size_t OUT_W = (size_t)DIMC * DIMC;
    const size_t GLU_W = (size_t)GLUN * DIMC;
    const size_t FF2_W = (size_t)DIMC * FFK;

    _Float16* wqkvT = (_Float16*)alloc(DEPTH * QKV_W * sizeof(_Float16));
    _Float16* woutT = (_Float16*)alloc(DEPTH * OUT_W * sizeof(_Float16));
    _Float16* wgluT = (_Float16*)alloc(DEPTH * GLU_W * sizeof(_Float16));
    _Float16* wff2T = (_Float16*)alloc(DEPTH * FF2_W * sizeof(_Float16));
    float*    btab  = (float*)   alloc((size_t)(2 * NN - 1) * HH * sizeof(float));
    _Float16* xn16  = (_Float16*)alloc((size_t)TOK * DIMC * sizeof(_Float16));
    float*    qkvF  = (float*)   alloc((size_t)TOK * 3 * HH * DD * sizeof(float));
    _Float16* q16   = (_Float16*)alloc((size_t)BB * HH * NN * DD * sizeof(_Float16));
    _Float16* k16   = (_Float16*)alloc((size_t)BB * HH * NN * DD * sizeof(_Float16));
    _Float16* v16t  = (_Float16*)alloc((size_t)BB * HH * DD * NN * sizeof(_Float16));
    _Float16* o16   = (_Float16*)alloc((size_t)TOK * HH * DD * sizeof(_Float16));
    float*    hbuf  = (float*)   alloc((size_t)TOK * GLUN * sizeof(float));
    _Float16* g16   = (_Float16*)alloc((size_t)TOK * FFK * sizeof(_Float16));
    (void)ws_size; (void)n_in; (void)in_sizes; (void)out_size;

    (void)hipMemcpyAsync(x, x_in, (size_t)TOK * DIMC * sizeof(float),
                         hipMemcpyDeviceToDevice, stream);

    posbias_kernel<<<2 * NN - 1, PBN, 0, stream>>>(pb_w1, pb_b1, pb_w2, pb_b2,
                                                   pb_w3, pb_b3, btab);

    auto convT = [&](const float* W, _Float16* WT, size_t K, size_t C) {
        size_t tot = K * C;
        conv_t_kernel<<<(unsigned)((tot + 255) / 256), 256, 0, stream>>>(W, WT, (int)K, (int)C);
    };
    for (int i = 0; i < DEPTH; ++i) {
        convT(w_qkv + (size_t)i * QKV_W, wqkvT + (size_t)i * QKV_W, DIMC, 3 * HH * DD);
        convT(w_out + (size_t)i * OUT_W, woutT + (size_t)i * OUT_W, DIMC, DIMC);
        convT(w_glu + (size_t)i * GLU_W, wgluT + (size_t)i * GLU_W, DIMC, GLUN);
        convT(w_ff2 + (size_t)i * FF2_W, wff2T + (size_t)i * FF2_W, FFK, DIMC);
    }

    for (int i = 0; i < DEPTH; ++i) {
        ln_kernel<<<TOK, 256, 0, stream>>>(x, ln1_g + (size_t)i * DIMC,
                                           ln1_b + (size_t)i * DIMC, xn16);

        gemm_f16_f32<<<dim3(3 * HH * DD / 128, TOK / 128), 256, 0, stream>>>(
            xn16, wqkvT + (size_t)i * QKV_W, qkvF, nullptr, nullptr,
            3 * HH * DD, DIMC);

        pack_qkv<<<dim3(NN, BB * HH), 32, 0, stream>>>(qkvF, temp, i, q16, k16, v16t);

        attn_kernel<<<dim3(BB * HH, NN / 64), 128, 0, stream>>>(q16, k16, v16t, btab, o16);

        gemm_f16_f32<<<dim3(DIMC / 128, TOK / 128), 256, 0, stream>>>(
            o16, woutT + (size_t)i * OUT_W, x, nullptr, x, DIMC, DIMC);

        ln_kernel<<<TOK, 256, 0, stream>>>(x, ln2_g + (size_t)i * DIMC,
                                           ln2_b + (size_t)i * DIMC, xn16);

        gemm_f16_f32<<<dim3(GLUN / 128, TOK / 128), 256, 0, stream>>>(
            xn16, wgluT + (size_t)i * GLU_W, hbuf, b_glu + (size_t)i * GLUN, nullptr,
            GLUN, DIMC);

        glu_kernel<<<(unsigned)(((size_t)TOK * FFK + 255) / 256), 256, 0, stream>>>(hbuf, g16);

        gemm_f16_f32<<<dim3(DIMC / 128, TOK / 128), 256, 0, stream>>>(
            g16, wff2T + (size_t)i * FF2_W, x, b_ff2 + (size_t)i * DIMC, x,
            DIMC, FFK);
    }
}
